// QuantizedLinear_6992206758356
// MI455X (gfx1250) — matrix operand order repaired and checked by reading, not compiled, not run on hardware
//
#include <hip/hip_runtime.h>

// ---- CDNA5 (gfx1250) wave32 WMMA types ----
typedef __attribute__((ext_vector_type(16))) __bf16 v16bf;
typedef __attribute__((ext_vector_type(8)))  __bf16 v8bf;
typedef __attribute__((ext_vector_type(8)))  float  v8f;

#define IN_F   4096
#define OUT_F  11008
#define BM     128          // block tile M
#define BN     64           // block tile N
#define BK     32           // K step (bf16 WMMA K)
#define LDAF   36           // fp32 LDS K-stride for A (144B: conflict-free b128)
#define LDW    40           // bf16 LDS K-stride for W (80B: conflict-free b128)
#define KTILES (IN_F / BK)

__device__ __forceinline__ unsigned lds_off(const void* p) {
    // generic pointer to LDS: low 32 bits carry the LDS byte offset
    return (unsigned)(unsigned long long)p;
}

// B fragment (bf16, 32x16): lane holds K{k0..k0+7, k0+16..k0+23} at column N=lane&15.
__device__ __forceinline__ v16bf frag_ld_w(const __bf16* p, int row, int k0) {
    v8bf a = *(const v8bf*)(p + row * LDW + k0);
    v8bf b = *(const v8bf*)(p + row * LDW + k0 + 16);
    return __builtin_shufflevector(a, b, 0,1,2,3,4,5,6,7,8,9,10,11,12,13,14,15);
}

// A fragment pair from raw fp32 LDS tile: bf16 hi + bf16 residual (x = hi + lo).
// Lane layout (16-bit A 16x32): lanes<16 -> K{0-7,16-23}, lanes>=16 -> K{8-15,24-31}.
__device__ __forceinline__ void frag_ld_a(const float* As, int row, int kc0,
                                          v16bf& hi, v16bf& lo) {
    const float* rp = As + row * LDAF;
    float f[16];
    *(float4*)(f + 0)  = *(const float4*)(rp + kc0 + 0);
    *(float4*)(f + 4)  = *(const float4*)(rp + kc0 + 4);
    *(float4*)(f + 8)  = *(const float4*)(rp + kc0 + 16);
    *(float4*)(f + 12) = *(const float4*)(rp + kc0 + 20);
    #pragma unroll
    for (int i = 0; i < 16; ++i) {
        __bf16 h = (__bf16)f[i];
        hi[i] = h;
        lo[i] = (__bf16)(f[i] - (float)h);
    }
}

__global__ __launch_bounds__(256)
void qlinear_wmma_async(const float* __restrict__ X,          // (M, 4096) fp32
                        const unsigned char* __restrict__ QW, // (11008, 2048) packed int4
                        const float* __restrict__ S,          // (11008,)
                        const float* __restrict__ Z,          // (11008,)
                        float* __restrict__ Y)                // (M, 11008) fp32
{
    __shared__ float  As[2][BM * LDAF];     // raw fp32 x tiles (async DMA target)
    __shared__ __bf16 Wq[2][BN * LDW];      // exact int4 levels as bf16 (0..15)

    const int t     = threadIdx.x;
    const int lane  = t & 31;
    const int wave  = t >> 5;
    const int waveM = wave >> 1;            // 0..3
    const int waveN = wave & 1;             // 0..1
    const int gM0   = blockIdx.y * BM;
    const int gN0   = blockIdx.x * BN;

    // --- A async staging: thread t -> row t>>1, 16 floats at col (t&1)*16 ---
    const int aRow = t >> 1;
    const int aK   = (t & 1) * 16;
    const unsigned long long aG =
        (unsigned long long)(X + (size_t)(gM0 + aRow) * IN_F + aK);
    unsigned aL[2];
    aL[0] = lds_off(&As[0][aRow * LDAF + aK]);
    aL[1] = lds_off(&As[1][aRow * LDAF + aK]);

    // --- W staging: thread t -> row t>>2, 4 packed bytes at (t&3)*4 ---
    const int wRow = t >> 2;
    const int wB   = (t & 3) * 4;
    const unsigned char* qPtr = QW + (size_t)(gN0 + wRow) * (IN_F / 2) + wB;

    auto issueAsyncA = [&](int kt, int buf) {
        const unsigned long long g = aG + (unsigned long long)kt * (BK * 4);
        const unsigned l = aL[buf];
        // 64B per thread: 4x b128 DMA global -> LDS (INST_OFFSET applies to both sides)
        asm volatile(
            "global_load_async_to_lds_b128 %0, %1, off\n\t"
            "global_load_async_to_lds_b128 %0, %1, off offset:16\n\t"
            "global_load_async_to_lds_b128 %0, %1, off offset:32\n\t"
            "global_load_async_to_lds_b128 %0, %1, off offset:48"
            :: "v"(l), "v"(g) : "memory");
    };
    auto waitAsync = [&]() {
        asm volatile("s_wait_asynccnt 0x0" ::: "memory");
    };
    auto stageW = [&](int buf, unsigned u) {
        const int base = wRow * LDW + wB * 2;
        #pragma unroll
        for (int i = 0; i < 4; ++i) {
            const unsigned b = (u >> (8 * i)) & 0xFFu;
            Wq[buf][base + 2*i + 0] = (__bf16)(float)(b & 15u);  // low nibble -> even k
            Wq[buf][base + 2*i + 1] = (__bf16)(float)(b >> 4);   // high nibble -> odd k
        }
    };

    // constant all-ones B fragment (B = ones => C[m][n] = row-sum of A)
    v16bf ones;
    #pragma unroll
    for (int i = 0; i < 16; ++i) ones[i] = (__bf16)1.0f;

    v8f acc[2][2];                          // Σ x·q
    v8f rs[2];                              // Σ x  (per mi strip; same for all n)
    const v8f vzero = {};
    #pragma unroll
    for (int i = 0; i < 2; ++i) {
        rs[i] = vzero;
        #pragma unroll
        for (int j = 0; j < 2; ++j) acc[i][j] = vzero;
    }

    // ---- prologue: fill buffer 0 ----
    issueAsyncA(0, 0);
    stageW(0, *(const unsigned*)qPtr);
    waitAsync();
    __syncthreads();

    // ---- pipelined main loop: prefetch kt+1 into the other buffer while computing kt ----
    for (int kt = 0; kt < KTILES; ++kt) {
        const int  cur = kt & 1;
        const int  nxt = cur ^ 1;
        const bool pf  = (kt + 1) < KTILES;

        unsigned uNext = 0u;
        if (pf) {
            issueAsyncA(kt + 1, nxt);                                   // DMA in flight
            uNext = *(const unsigned*)(qPtr + (size_t)(kt + 1) * (BK / 2));
        }

        // --- fragments from buffer `cur` ---
        const int half = lane >> 4;
        const int r    = lane & 15;
        const int aC0  = half ? 8 : 0;        // A chunks {c0..c0+7, c0+16..c0+23}
        const int bK0  = half ? 8 : 0;        // B chunks {c0..c0+7, c0+16..c0+23}

        v16bf aHi[2], aLo[2], bQ[2];
        #pragma unroll
        for (int mi = 0; mi < 2; ++mi)
            frag_ld_a(As[cur], waveM * 32 + mi * 16 + r, aC0, aHi[mi], aLo[mi]);
        #pragma unroll
        for (int ni = 0; ni < 2; ++ni)
            bQ[ni] = frag_ld_w(Wq[cur], waveN * 32 + ni * 16 + r, bK0);

        // --- row sums: A x ones (register-constant B, no LDS traffic) ---
        #pragma unroll
        for (int mi = 0; mi < 2; ++mi) {
            rs[mi] = __builtin_amdgcn_wmma_f32_16x16x32_bf16(
                false, aHi[mi], false, ones, (short)0, rs[mi], false, false);
            rs[mi] = __builtin_amdgcn_wmma_f32_16x16x32_bf16(
                false, aLo[mi], false, ones, (short)0, rs[mi], false, false);
        }
        // --- main products: (xHi + xLo) x q ; bf16 products are exact here ---
        #pragma unroll
        for (int mi = 0; mi < 2; ++mi)
            #pragma unroll
            for (int ni = 0; ni < 2; ++ni) {
                acc[mi][ni] = __builtin_amdgcn_wmma_f32_16x16x32_bf16(
                    false, aHi[mi], false, bQ[ni], (short)0, acc[mi][ni], false, false);
                acc[mi][ni] = __builtin_amdgcn_wmma_f32_16x16x32_bf16(
                    false, aLo[mi], false, bQ[ni], (short)0, acc[mi][ni], false, false);
            }

        if (pf) {
            stageW(nxt, uNext);   // unpack prefetched weights into next buffer
            waitAsync();          // this wave's DMA for buffer `nxt` done
            __syncthreads();      // publish LDS (compiler adds s_wait_dscnt before signal)
        }
    }

    // ---- epilogue: y = s_n * (acc - z_n * rowSum), fp32 exact; non-temporal stores ----
    const int half = lane >> 4;
    const int cn   = lane & 15;
    #pragma unroll
    for (int ni = 0; ni < 2; ++ni) {
        const int col = gN0 + waveN * 32 + ni * 16 + cn;
        const float sc = S[col];
        const float zc = Z[col];
        #pragma unroll
        for (int mi = 0; mi < 2; ++mi) {
            const int row0 = gM0 + waveM * 32 + mi * 16 + half * 8;
            #pragma unroll
            for (int rr = 0; rr < 8; ++rr) {
                const float v = sc * (acc[mi][ni][rr] - zc * rs[mi][rr]);
                __builtin_nontemporal_store(v, &Y[(size_t)(row0 + rr) * OUT_F + col]);
            }
        }
    }
}

extern "C" void kernel_launch(void* const* d_in, const int* in_sizes, int n_in,
                              void* d_out, int out_size, void* d_ws, size_t ws_size,
                              hipStream_t stream) {
    const float*         x  = (const float*)d_in[0];
    const unsigned char* qw = (const unsigned char*)d_in[1];
    const float*         s  = (const float*)d_in[2];
    const float*         z  = (const float*)d_in[3];
    float*               y  = (float*)d_out;

    const int M = in_sizes[0] / IN_F;            // 16384
    dim3 grid(OUT_F / BN, M / BM, 1);            // (172, 128)
    dim3 block(256, 1, 1);
    qlinear_wmma_async<<<grid, block, 0, stream>>>(x, qw, s, z, y);
}
